// Model_6210522710554
// MI455X (gfx1250) — compile-verified, matrix operand order checked by reading, not run on hardware
//
#include <hip/hip_runtime.h>
#include <stddef.h>

// ---------------------------------------------------------------------------
// Problem dims (fixed by the reference)
// ---------------------------------------------------------------------------
#define Bn 512
#define Tn 200
#define Vn 2048
#define En 20
#define Hn 100
#define In 64
#define An 32
#define G4H 400           // 4*H
#define NT_G 25           // gate n-tiles (400/16)
#define BT 32             // batch tiles (512/16)

typedef __attribute__((ext_vector_type(16))) __bf16 bf16x16;
typedef __attribute__((ext_vector_type(8)))  __bf16 bf16x8;
typedef __attribute__((ext_vector_type(8)))  float  f32x8;
typedef __attribute__((ext_vector_type(4)))  float  f32x4;

// ---------------------------------------------------------------------------
// Workspace layout (bytes). Total ~165 MB.
// ---------------------------------------------------------------------------
static constexpr size_t OFF_WEMB = 0;                                   // bf16 [32][2048]  (E padded 20->32, pad rows = 0)
static constexpr size_t OFF_WIH  = OFF_WEMB + (size_t)32 * 2048 * 2;    // bf16 [400][32]   (K padded 20->32)
static constexpr size_t OFF_WHH  = OFF_WIH  + (size_t)400 * 32 * 2;     // bf16 [400][128]  (K padded 100->128)
static constexpr size_t OFF_W1   = OFF_WHH  + (size_t)400 * 128 * 2;    // bf16 [64][128]
static constexpr size_t OFF_W2   = OFF_W1   + (size_t)64 * 128 * 2;     // bf16 [32][64]
static constexpr size_t OFF_BIAS = OFF_W2   + (size_t)32 * 64 * 2;      // f32  [400]  (b_ih + b_hh)
static constexpr size_t OFF_HFIN = (OFF_BIAS + (size_t)400 * 4 + 255) & ~(size_t)255;  // bf16 [512][128]
static constexpr size_t OFF_XG   = (OFF_HFIN + (size_t)512 * 128 * 2 + 255) & ~(size_t)255;
// xg: f32 fragment-layout tiles [T][BT][NT_G][lane=32][8]  = 200*32*25*256 floats = 163.84 MB

// ---------------------------------------------------------------------------
// WMMA helpers (CDNA5 16x16x32 bf16, f32 accumulate)
// ---------------------------------------------------------------------------
__device__ __forceinline__ f32x8 wmma_bf16(bf16x16 a, bf16x16 b, f32x8 c) {
  // 8-arg form: (neg_a, A, neg_b, B, c_mod, C, reuse_a, reuse_b)
  return __builtin_amdgcn_wmma_f32_16x16x32_bf16(false, a, false, b, (short)0, c,
                                                 false, false);
}

// A/B 16-bit fragment, per-lane: elements 0..7 = K at p[0..7], 8..15 = K at p[16..23].
// Caller passes p = row_base + k0 + lo  where lo = (lane<16 ? 0 : 8).
__device__ __forceinline__ bf16x16 frag_bf16(const __bf16* p) {
  bf16x8 a = *(const bf16x8*)(p);
  bf16x8 b = *(const bf16x8*)(p + 16);
  bf16x16 r;
#pragma unroll
  for (int i = 0; i < 8; ++i) { r[i] = a[i]; r[i + 8] = b[i]; }
  return r;
}

__device__ __forceinline__ bf16x16 frag_from_f32(const float* p) {
  f32x4 a0 = *(const f32x4*)(p);
  f32x4 a1 = *(const f32x4*)(p + 4);
  f32x4 b0 = *(const f32x4*)(p + 16);
  f32x4 b1 = *(const f32x4*)(p + 20);
  bf16x16 r;
#pragma unroll
  for (int i = 0; i < 4; ++i) {
    r[i]      = (__bf16)a0[i];
    r[i + 4]  = (__bf16)a1[i];
    r[i + 8]  = (__bf16)b0[i];
    r[i + 12] = (__bf16)b1[i];
  }
  return r;
}

__device__ __forceinline__ f32x8 bcast8(float x) {
  f32x8 r;
#pragma unroll
  for (int v = 0; v < 8; ++v) r[v] = x;
  return r;
}

__device__ __forceinline__ float sigmoidf_(float x) {
  return 1.0f / (1.0f + __expf(-x));
}

// ---------------------------------------------------------------------------
// Kernel 0: pack/convert weights to padded bf16, sum biases, zero h_final pad
// ---------------------------------------------------------------------------
__global__ void prep_kernel(const float* __restrict__ W_emb, const float* __restrict__ W_ih,
                            const float* __restrict__ W_hh, const float* __restrict__ b_ih,
                            const float* __restrict__ b_hh, const float* __restrict__ W1,
                            const float* __restrict__ W2, char* __restrict__ ws) {
  __bf16* wembp = (__bf16*)(ws + OFF_WEMB);
  __bf16* wihp  = (__bf16*)(ws + OFF_WIH);
  __bf16* whhp  = (__bf16*)(ws + OFF_WHH);
  __bf16* w1p   = (__bf16*)(ws + OFF_W1);
  __bf16* w2p   = (__bf16*)(ws + OFF_W2);
  float*  biasg = (float*) (ws + OFF_BIAS);
  __bf16* hfin  = (__bf16*)(ws + OFF_HFIN);

  int idx = blockIdx.x * blockDim.x + threadIdx.x;
  int stride = gridDim.x * blockDim.x;

  for (int i = idx; i < 32 * 2048; i += stride) {          // W_emb^T cols, rows padded 20->32
    int r = i >> 11, c = i & 2047;
    wembp[i] = (__bf16)((r < En) ? W_emb[r * Vn + c] : 0.0f);
  }
  for (int i = idx; i < 400 * 32; i += stride) {           // W_ih, K padded 20->32
    int r = i >> 5, c = i & 31;
    wihp[i] = (__bf16)((c < En) ? W_ih[r * En + c] : 0.0f);
  }
  for (int i = idx; i < 400 * 128; i += stride) {          // W_hh, K padded 100->128
    int r = i >> 7, c = i & 127;
    whhp[i] = (__bf16)((c < Hn) ? W_hh[r * Hn + c] : 0.0f);
  }
  for (int i = idx; i < 64 * 128; i += stride) {           // W1, K padded 100->128
    int r = i >> 7, c = i & 127;
    w1p[i] = (__bf16)((c < Hn) ? W1[r * Hn + c] : 0.0f);
  }
  for (int i = idx; i < 32 * 64; i += stride) w2p[i] = (__bf16)W2[i];
  for (int i = idx; i < G4H; i += stride) biasg[i] = b_ih[i] + b_hh[i];
  for (int i = idx; i < 512 * 128; i += stride) hfin[i] = (__bf16)0.0f;  // zero (incl. K pad)
}

// ---------------------------------------------------------------------------
// Kernel 1: fused embedding + gate-precompute GEMM.
//   One wave per 16-row tile (16 batch rows @ fixed t). Stage 1: emb[16,32pad]
//   over K=2048 (bf16 WMMA). Stage 2: xg[16,400] = emb @ W_ih^T + (b_ih+b_hh),
//   stored in WMMA C-fragment layout for kernel 2.
// ---------------------------------------------------------------------------
__global__ __launch_bounds__(256) void embed_xg_kernel(const float* __restrict__ batch,
                                                       const float* __restrict__ b_emb,
                                                       char* __restrict__ ws) {
  __shared__ float sEmb[8][16][32];   // per-wave emb staging (C-layout -> A-layout)

  const int lane = threadIdx.x & 31;
  const int wave = threadIdx.x >> 5;
  const int gw = blockIdx.x * 8 + wave;       // 0 .. 6399 tiles
  const int t  = gw >> 5;                     // time step
  const int bt = gw & 31;                     // batch tile
  const int b0 = bt * 16;

  const int col = lane & 15;                  // lane row (A) / lane col (B,C)
  const int hi  = lane >> 4;
  const int lo  = hi ? 8 : 0;
  const int mbase = 8 * hi;

  const __bf16* wemb  = (const __bf16*)(ws + OFF_WEMB);
  const __bf16* wih   = (const __bf16*)(ws + OFF_WIH);
  const float*  biasg = (const float*) (ws + OFF_BIAS);
  float*        xg    = (float*)       (ws + OFF_XG);

  // ---- Stage 1: emb = batch_tile @ W_emb^T + b_emb (N padded to 32) ----
  const float* arow = batch + ((size_t)(b0 + col) * Tn + t) * Vn;   // A row for this lane
  const __bf16* brow0 = wemb + (size_t)col * 2048;                  // B col n = col
  const __bf16* brow1 = wemb + (size_t)(16 + col) * 2048;           // B col n = 16+col (pad rows are 0)

  f32x8 acc0 = bcast8(b_emb[col]);
  f32x8 acc1 = bcast8((16 + col) < En ? b_emb[16 + col] : 0.0f);

  for (int k0 = 0; k0 < Vn; k0 += 32) {
    bf16x16 a  = frag_from_f32(arow + k0 + lo);    // fp32 load + cvt to bf16 frag
    bf16x16 f0 = frag_bf16(brow0 + k0 + lo);
    bf16x16 f1 = frag_bf16(brow1 + k0 + lo);
    acc0 = wmma_bf16(a, f0, acc0);
    acc1 = wmma_bf16(a, f1, acc1);
  }

  // C-layout -> row-major staging in LDS (wave-private slab)
#pragma unroll
  for (int v = 0; v < 8; ++v) {
    sEmb[wave][mbase + v][col]      = acc0[v];
    sEmb[wave][mbase + v][16 + col] = acc1[v];
  }
  __syncthreads();

  // Re-read as A fragment (row = col lane, K = 0..31)
  bf16x16 afrag = frag_from_f32(&sEmb[wave][col][lo]);

  // ---- Stage 2: xg = emb @ W_ih^T + bias, one WMMA per 16-col tile ----
  for (int nt = 0; nt < NT_G; ++nt) {
    const int n = nt * 16 + col;
    f32x8 acc = bcast8(biasg[n]);
    bf16x16 bfr = frag_bf16(wih + (size_t)n * 32 + lo);
    acc = wmma_bf16(afrag, bfr, acc);
    // store in fragment layout: tile base + lane*8, fully coalesced b128 x2
    float* dst = xg + ((((size_t)t * BT + bt) * NT_G + nt) * 256) + (size_t)lane * 8;
    *(f32x8*)dst = acc;
  }
}

// ---------------------------------------------------------------------------
// Kernel 2: LSTM recurrence. One block = 16 batch rows, iterates all T=200
//   steps with h (bf16) / c (f32) resident in LDS. No grid sync needed.
//
//   Per wave: W_hh B-fragments for its 3-4 n-tiles are loaded ONCE into
//   registers (<=128 VGPRs) and held for all 200 steps. xg C-fragments are
//   register double-buffered (load t+1 while computing t). A-fragments of h
//   are hoisted per k-step; WMMAs iterate over independent accumulators to
//   avoid D->A RAW hazard NOPs.
// ---------------------------------------------------------------------------
__global__ __launch_bounds__(256, 1) void lstm_kernel(const int* __restrict__ lengths,
                                                      const float* __restrict__ h0,
                                                      const float* __restrict__ c0,
                                                      char* __restrict__ ws) {
  __shared__ float  sGates[16 * G4H];   // 25.6 KB
  __shared__ __bf16 sH[16 * 128];       // 4 KB (K padded, pad = 0)
  __shared__ float  sC[16 * Hn];        // 6.4 KB
  __shared__ int    sLen[16];

  const int bt = blockIdx.x;
  const int b0 = bt * 16;
  const int tid = threadIdx.x;
  const int lane = tid & 31;
  const int wave = tid >> 5;
  const int col = lane & 15;
  const int hi  = lane >> 4;
  const int lo  = hi ? 8 : 0;
  const int mbase = 8 * hi;

  const __bf16* whh = (const __bf16*)(ws + OFF_WHH);
  const float*  xg  = (const float*) (ws + OFF_XG);
  __bf16*       hfin = (__bf16*)(ws + OFF_HFIN);

  // n-tiles owned by this wave: nt = wave + 8*i  (wave 0 gets 4, others 3)
  const int myTiles = (wave == 0) ? 4 : 3;

  // ---- load W_hh B-fragments once; resident in VGPRs for all 200 steps ----
  bf16x16 bw[4][4];
#pragma unroll
  for (int i = 0; i < 4; ++i) {
    if (i < myTiles) {
      const int n = (wave + 8 * i) * 16 + col;
      const __bf16* brow = whh + (size_t)n * 128;
#pragma unroll
      for (int kk = 0; kk < 4; ++kk) bw[i][kk] = frag_bf16(brow + kk * 32 + lo);
    }
  }

  // init state
  for (int i = tid; i < 16 * Hn; i += 256) {
    int m = i / Hn, j = i - m * Hn;
    sC[i] = c0[(size_t)(b0 + m) * Hn + j];
  }
  for (int i = tid; i < 16 * 128; i += 256) {
    int m = i >> 7, j = i & 127;
    sH[i] = (__bf16)((j < Hn) ? h0[(size_t)(b0 + m) * Hn + j] : 0.0f);
  }
  if (tid < 16) sLen[tid] = lengths[b0 + tid] - 1;

  // ---- prime the xg double buffer with step 0 ----
  f32x8 nxt[4];
#pragma unroll
  for (int i = 0; i < 4; ++i)
    if (i < myTiles)
      nxt[i] = *(const f32x8*)(xg + (((size_t)bt * NT_G + (wave + 8 * i)) * 256) +
                               (size_t)lane * 8);
  __syncthreads();

  for (int t = 0; t < Tn; ++t) {
    // accumulator init = xg fragment (already in registers)
    f32x8 acc[4];
#pragma unroll
    for (int i = 0; i < 4; ++i)
      if (i < myTiles) acc[i] = nxt[i];

    // kick off next step's xg fragment loads (latency hidden behind WMMAs)
    if (t + 1 < Tn) {
#pragma unroll
      for (int i = 0; i < 4; ++i)
        if (i < myTiles)
          nxt[i] = *(const f32x8*)(xg +
                    ((((size_t)(t + 1) * BT + bt) * NT_G + (wave + 8 * i)) * 256) +
                    (size_t)lane * 8);
    }

    // h A-fragments: one per k-step, shared across all owned n-tiles
    bf16x16 a[4];
#pragma unroll
    for (int kk = 0; kk < 4; ++kk) a[kk] = frag_bf16(&sH[col * 128 + kk * 32 + lo]);

    // independent accumulator chains: kk outer, tile inner
#pragma unroll
    for (int kk = 0; kk < 4; ++kk) {
#pragma unroll
      for (int i = 0; i < 4; ++i)
        if (i < myTiles) acc[i] = wmma_bf16(a[kk], bw[i][kk], acc[i]);
    }

    // stage gates to LDS (C-layout -> row-major)
#pragma unroll
    for (int i = 0; i < 4; ++i) {
      if (i < myTiles) {
        const int n = (wave + 8 * i) * 16 + col;
#pragma unroll
        for (int v = 0; v < 8; ++v) sGates[(mbase + v) * G4H + n] = acc[i][v];
      }
    }
    __syncthreads();

    // --- elementwise gate update: 16*100 = 1600 work items ---
    for (int i = tid; i < 16 * Hn; i += 256) {
      int m = i / Hn, j = i - m * Hn;
      const float* g = &sGates[m * G4H];
      float ig = g[j], fg = g[Hn + j], gg = g[2 * Hn + j], og = g[3 * Hn + j];
      float cc = sigmoidf_(fg) * sC[i] + sigmoidf_(ig) * tanhf(gg);
      float hh = sigmoidf_(og) * tanhf(cc);
      sC[i] = cc;
      sH[m * 128 + j] = (__bf16)hh;
      if (t == sLen[m]) hfin[(size_t)(b0 + m) * 128 + j] = (__bf16)hh;
    }
    __syncthreads();
  }
}

// ---------------------------------------------------------------------------
// Kernel 3: MLP head. One wave per 16 rows: relu(hf @ W1^T + b1) @ W2^T + b2.
// ---------------------------------------------------------------------------
__global__ __launch_bounds__(32) void head_kernel(const float* __restrict__ b1,
                                                  const float* __restrict__ b2,
                                                  const char* __restrict__ ws,
                                                  float* __restrict__ out) {
  __shared__ __bf16 sInter[16 * 64];   // 2 KB

  const int b0 = blockIdx.x * 16;
  const int lane = threadIdx.x & 31;
  const int col = lane & 15;
  const int hi  = lane >> 4;
  const int lo  = hi ? 8 : 0;
  const int mbase = 8 * hi;

  const __bf16* hf = (const __bf16*)(ws + OFF_HFIN);
  const __bf16* w1 = (const __bf16*)(ws + OFF_W1);
  const __bf16* w2 = (const __bf16*)(ws + OFF_W2);

  // Stage 1: inter = relu(hf @ W1^T + b1)   M=16, N=64, K=128(pad)
  for (int nt = 0; nt < 4; ++nt) {
    const int n = nt * 16 + col;
    f32x8 acc = bcast8(b1[n]);
    const __bf16* brow = w1 + (size_t)n * 128;
#pragma unroll
    for (int kk = 0; kk < 4; ++kk) {
      const int k0 = kk * 32;
      bf16x16 a = frag_bf16(hf + (size_t)(b0 + col) * 128 + k0 + lo);
      bf16x16 b = frag_bf16(brow + k0 + lo);
      acc = wmma_bf16(a, b, acc);
    }
#pragma unroll
    for (int v = 0; v < 8; ++v)
      sInter[(mbase + v) * 64 + n] = (__bf16)fmaxf(acc[v], 0.0f);
  }
  __syncthreads();

  // Stage 2: out = inter @ W2^T + b2   M=16, N=32, K=64
  for (int nt = 0; nt < 2; ++nt) {
    const int n = nt * 16 + col;
    f32x8 acc = bcast8(b2[n]);
    const __bf16* brow = w2 + (size_t)n * 64;
#pragma unroll
    for (int kk = 0; kk < 2; ++kk) {
      const int k0 = kk * 32;
      bf16x16 a = frag_bf16(&sInter[col * 64 + k0 + lo]);
      bf16x16 b = frag_bf16(brow + k0 + lo);
      acc = wmma_bf16(a, b, acc);
    }
#pragma unroll
    for (int v = 0; v < 8; ++v)
      out[(size_t)(b0 + mbase + v) * An + n] = acc[v];
  }
}

// ---------------------------------------------------------------------------
// Launcher. Inputs in setup_inputs() order. Needs ws_size >= ~165 MB
// (xg fragment buffer dominates: 163.84 MB).
// ---------------------------------------------------------------------------
extern "C" void kernel_launch(void* const* d_in, const int* in_sizes, int n_in,
                              void* d_out, int out_size, void* d_ws, size_t ws_size,
                              hipStream_t stream) {
  const float* batch   = (const float*)d_in[0];
  const int*   lengths = (const int*)  d_in[1];
  const float* W_emb   = (const float*)d_in[2];
  const float* b_emb   = (const float*)d_in[3];
  const float* W_ih    = (const float*)d_in[4];
  const float* W_hh    = (const float*)d_in[5];
  const float* b_ih    = (const float*)d_in[6];
  const float* b_hh    = (const float*)d_in[7];
  const float* W1      = (const float*)d_in[8];
  const float* b1      = (const float*)d_in[9];
  const float* W2      = (const float*)d_in[10];
  const float* b2      = (const float*)d_in[11];
  const float* h0      = (const float*)d_in[12];
  const float* c0      = (const float*)d_in[13];
  char* ws = (char*)d_ws;
  float* out = (float*)d_out;

  prep_kernel<<<512, 256, 0, stream>>>(W_emb, W_ih, W_hh, b_ih, b_hh, W1, W2, ws);
  embed_xg_kernel<<<(Bn / 16) * Tn / 8, 256, 0, stream>>>(batch, b_emb, ws);
  lstm_kernel<<<Bn / 16, 256, 0, stream>>>(lengths, h0, c0, ws);
  head_kernel<<<Bn / 16, 32, 0, stream>>>(b1, b2, ws, out);
}
